// SelfAttention_42494406426644
// MI455X (gfx1250) — matrix operand order repaired and checked by reading, not compiled, not run on hardware
//
#include <hip/hip_runtime.h>

typedef _Float16 v16h __attribute__((ext_vector_type(16)));
typedef _Float16 h8   __attribute__((ext_vector_type(8)));
typedef _Float16 h4   __attribute__((ext_vector_type(4)));
typedef float    v8f  __attribute__((ext_vector_type(8)));
typedef int      i32x4 __attribute__((vector_size(16)));   // matches builtin proto

constexpr int kB = 2, kL = 2048, kD = 2048, kH = 32, kHD = 64;
constexpr int kM = kB * kL;   // 4096 rows in both GEMMs

// ---------------------------------------------------------------------------
// CDNA5 async global->LDS copy (ASYNCcnt path), with safe fallback.
// Toolchain prototype (from hipcc diagnostic): parameters are generic
// pointers to 16-byte int vectors; source-global first (matches the
// __builtin_amdgcn_global_load_lds / load_to_lds family ordering).
// ---------------------------------------------------------------------------
#if defined(__gfx1250__) &&                                              \
    __has_builtin(__builtin_amdgcn_global_load_async_to_lds_b128) &&     \
    __has_builtin(__builtin_amdgcn_s_wait_asynccnt)
#define USE_ASYNC_LDS 1
#else
#define USE_ASYNC_LDS 0
#endif

static __device__ inline void cp16_async(void* lds, const void* g) {
#if USE_ASYNC_LDS
  __builtin_amdgcn_global_load_async_to_lds_b128(
      (i32x4*)g, (i32x4*)lds, /*offset=*/0, /*cpol=*/0);
#else
  *(uint4*)lds = *(const uint4*)g;
#endif
}
static __device__ inline void cp_wait_all() {
#if USE_ASYNC_LDS
  __builtin_amdgcn_s_wait_asynccnt(0);
#endif
}

// Build a v16h A/B fragment from two 16-byte chunks (per the gfx1250
// 16-bit WMMA VGPR layouts: each lane's 16 halves are two contiguous
// 8-half groups).
static __device__ inline v16h frag16(const _Float16* p0, const _Float16* p1) {
  h8 lo = *(const h8*)p0;
  h8 hi = *(const h8*)p1;
  v16h r;
#pragma unroll
  for (int i = 0; i < 8; ++i) { r[i] = lo[i]; r[i + 8] = hi[i]; }
  return r;
}

// ---------------------------------------------------------------------------
// Generic f32-in / f32-out GEMM:  out[m,n] = sum_k A[m,k] * W[n,k]
// Block: 256 threads (8 waves), tile 128(M) x 128(N), K staged 32 at a time
// as f16 in double-buffered LDS (one barrier per k-step; staging of slab
// k+32 overlaps WMMA on slab k). Wave grid 2x4; each wave computes 64x32.
// MODE 0: out is [M x N] row-major.
// MODE 1: out is qkv workspace [3][B][H][L][HD] (QKV epilogue scatter).
// ---------------------------------------------------------------------------
template <int MODE>
__global__ __launch_bounds__(256) void gemm_wmma(
    const float* __restrict__ A, const float* __restrict__ W,
    float* __restrict__ out, int N, int K) {
  __shared__ _Float16 As[2][128 * 40];  // padded row stride 40 halves
  __shared__ _Float16 Bs[2][128 * 40];

  const int tid  = threadIdx.x;
  const int wid  = tid >> 5;
  const int lane = tid & 31;
  const int wm   = wid >> 2;       // 0..1  (M)
  const int wn   = wid & 3;        // 0..3  (N)
  const int lrow = lane & 15;
  const bool hiL = lane >= 16;

  const int srow = tid >> 1;       // staging row 0..127
  const int sseg = tid & 1;        // which 16-float half of the 32-k slab
  const size_t aRowOff = (size_t)(blockIdx.y * 128 + srow) * K;
  const size_t wRowOff = (size_t)(blockIdx.x * 128 + srow) * K;

  auto stage = [&](int buf, int k0) {
    const float4* ap = (const float4*)(A + aRowOff + k0 + sseg * 16);
    const float4* wp = (const float4*)(W + wRowOff + k0 + sseg * 16);
#pragma unroll
    for (int j = 0; j < 4; ++j) {
      float4 fa = ap[j];
      h4 ha = {(_Float16)fa.x, (_Float16)fa.y, (_Float16)fa.z, (_Float16)fa.w};
      *(h4*)&As[buf][srow * 40 + sseg * 16 + j * 4] = ha;
      float4 fw = wp[j];
      h4 hw = {(_Float16)fw.x, (_Float16)fw.y, (_Float16)fw.z, (_Float16)fw.w};
      *(h4*)&Bs[buf][srow * 40 + sseg * 16 + j * 4] = hw;
    }
  };

  v8f acc[4][2];
  const v8f vzero = {0.f, 0.f, 0.f, 0.f, 0.f, 0.f, 0.f, 0.f};
#pragma unroll
  for (int mt = 0; mt < 4; ++mt)
#pragma unroll
    for (int nt = 0; nt < 2; ++nt) acc[mt][nt] = vzero;

  stage(0, 0);

  for (int k0 = 0; k0 < K; k0 += 32) {
    const int cur = (k0 >> 5) & 1;
    __syncthreads();  // staged 'cur' visible; prev compute done before restage
    if (k0 + 32 < K) {
      if (k0 + 64 < K) {  // keep L2->WGP ahead of the staging loads
        __builtin_prefetch(A + aRowOff + k0 + 64 + sseg * 16, 0, 1);
        __builtin_prefetch(W + wRowOff + k0 + 64 + sseg * 16, 0, 1);
      }
      stage(cur ^ 1, k0 + 32);
    }

    // ---- fragments + 8 WMMAs from buffer 'cur' ------------------------------
    v16h af[4], bf[2];
#pragma unroll
    for (int mt = 0; mt < 4; ++mt) {
      const int r  = wm * 64 + mt * 16 + lrow;
      const int cb = hiL ? 8 : 0;  // A-frag: lanes16-31 hold k 8..15 / 24..31
      af[mt] = frag16(&As[cur][r * 40 + cb], &As[cur][r * 40 + cb + 16]);
    }
#pragma unroll
    for (int nt = 0; nt < 2; ++nt) {
      const int r  = wn * 32 + nt * 16 + lrow;
      const int cb = hiL ? 8 : 0;  // B-frag: lanes16-31 hold k 8..15 / 24..31
      bf[nt] = frag16(&Bs[cur][r * 40 + cb], &Bs[cur][r * 40 + cb + 16]);
    }
#pragma unroll
    for (int mt = 0; mt < 4; ++mt)
#pragma unroll
      for (int nt = 0; nt < 2; ++nt)
        acc[mt][nt] = __builtin_amdgcn_wmma_f32_16x16x32_f16(
            false, af[mt], false, bf[nt], (short)0, acc[mt][nt], false, false);
  }

  // ---- epilogue -------------------------------------------------------------
#pragma unroll
  for (int mt = 0; mt < 4; ++mt)
#pragma unroll
    for (int nt = 0; nt < 2; ++nt)
#pragma unroll
      for (int r = 0; r < 8; ++r) {
        const int grow = blockIdx.y * 128 + wm * 64 + mt * 16 + r + (hiL ? 8 : 0);
        const int gcol = blockIdx.x * 128 + wn * 32 + nt * 16 + lrow;
        const float v = acc[mt][nt][r];
        if (MODE == 0) {
          out[(size_t)grow * N + gcol] = v;
        } else {
          // grow -> (b, l); gcol -> (which of q/k/v, head, hd)
          const int b = grow >> 11, l = grow & (kL - 1);
          const int which = gcol >> 11;          // / 2048
          const int rem   = gcol & (kD - 1);
          const int head  = rem >> 6, hd = rem & (kHD - 1);
          out[((((size_t)which * kB + b) * kH + head) * kL + l) * kHD + hd] = v;
        }
      }
}

// ---------------------------------------------------------------------------
// RoPE on q,k + pack to f16. q is pre-scaled by 1/sqrt(HD)=0.125 (exact in
// f16). v written transposed [B][H][HD][L] so attention's P*V B-fragments are
// contiguous. One block = 4 (b,h,l) rows, 64 lanes per row (d = 0..63).
// ---------------------------------------------------------------------------
__global__ __launch_bounds__(256) void rope_pack(
    const float* __restrict__ qkv, const float* __restrict__ cosT,
    const float* __restrict__ sinT, _Float16* __restrict__ qh,
    _Float16* __restrict__ kh, _Float16* __restrict__ vth) {
  const int tid = threadIdx.x;
  const int lr = tid >> 6, d = tid & 63;
  const long g = (long)blockIdx.x * 4 + lr;       // (b*H + h)*L + l
  const int l = (int)(g % kL);
  const long bh = g / kL;
  const int h = (int)(bh % kH), b = (int)(bh / kH);

  const size_t base = (((size_t)b * kH + h) * kL + l) * kHD;
  const size_t qkvStride = (size_t)kB * kH * kL * kHD;
  const int d2 = (d < 32) ? d + 32 : d - 32;

  const float q  = qkv[base + d];
  const float k  = qkv[qkvStride + base + d];
  const float v  = qkv[2 * qkvStride + base + d];
  const float qp = qkv[base + d2];
  const float kp = qkv[qkvStride + base + d2];
  const float c  = cosT[((size_t)b * kL + l) * kHD + d];
  const float s  = sinT[((size_t)b * kL + l) * kHD + d];

  const float qo = (d < 32) ? (q * c - qp * s) : (q * c + qp * s);
  const float ko = (d < 32) ? (k * c - kp * s) : (k * c + kp * s);

  qh[base + d] = (_Float16)(qo * 0.125f);
  kh[base + d] = (_Float16)ko;
  vth[(((size_t)b * kH + h) * kHD + d) * kL + l] = (_Float16)v;
}

// ---------------------------------------------------------------------------
// Flash attention, causal. Block = 128 threads (4 waves) per (b, h, 64-query
// block). Each wave owns 16 queries: Q A-frags held in registers, K / V^T
// tiles (64 keys) staged in padded LDS via async global->LDS copies, online
// softmax with 16-lane shuffle reductions (C-tile rows live in one 16-lane
// half on wave32).
// ---------------------------------------------------------------------------
__global__ __launch_bounds__(128) void attn_wmma(
    const _Float16* __restrict__ qh, const _Float16* __restrict__ kh,
    const _Float16* __restrict__ vth, float* __restrict__ yws) {
  __shared__ _Float16 Ks[64 * 72];       // [key][hd],  padded stride 72
  __shared__ _Float16 Vs[64 * 72];       // [hd][key],  padded stride 72
  __shared__ _Float16 Ps[4 * 16 * 72];   // per-wave P tile [row][key]

  const int tid  = threadIdx.x;
  const int wave = tid >> 5;
  const int lane = tid & 31;
  const int lrow = lane & 15;
  const bool hiL = lane >= 16;
  const int qblk = blockIdx.x, h = blockIdx.y, b = blockIdx.z;

  const size_t headRow = ((size_t)b * kH + h) * kL;

  // Q fragments (16 queries x 64 hd), persistent in registers.
  const int qrow = qblk * 64 + wave * 16 + lrow;
  const _Float16* qp = qh + (headRow + qrow) * kHD;
  v16h qa[2];
#pragma unroll
  for (int c = 0; c < 2; ++c) {
    const int off = c * 32 + (hiL ? 8 : 0);
    qa[c] = frag16(qp + off, qp + off + 16);
  }

  const v8f vzero = {0.f, 0.f, 0.f, 0.f, 0.f, 0.f, 0.f, 0.f};
  float m_s[8], l_s[8];
  v8f yacc[4];
#pragma unroll
  for (int r = 0; r < 8; ++r) { m_s[r] = -1e30f; l_s[r] = 0.f; }
#pragma unroll
  for (int n = 0; n < 4; ++n) yacc[n] = vzero;

  for (int kb = 0; kb <= qblk; ++kb) {
    // ---- async tile staging (K block is 8KB contiguous in global) ----------
    const char* kg = (const char*)(kh + (headRow + (size_t)kb * 64) * kHD);
#pragma unroll
    for (int j = 0; j < 4; ++j) {
      const int cix = j * 128 + tid;           // 16B chunk index, coalesced
      const int key = cix >> 3, col = (cix & 7) * 8;
      cp16_async(&Ks[key * 72 + col], kg + (size_t)cix * 16);
    }
    {
      const int row = tid >> 1, part = tid & 1;  // row = hd
      const char* vg = (const char*)(vth + (((size_t)b * kH + h) * kHD + row) * kL +
                                     (size_t)kb * 64 + part * 32);
#pragma unroll
      for (int j = 0; j < 4; ++j)
        cp16_async(&Vs[row * 72 + part * 32 + j * 8], vg + (size_t)j * 16);
    }
    cp_wait_all();
    __syncthreads();

    // ---- S = (Q/8) * K^T : 4 key tiles x 2 k-chunks = 8 WMMA ----------------
    v8f s[4];
#pragma unroll
    for (int n = 0; n < 4; ++n) {
      const int key = n * 16 + lrow;
      const _Float16* kr = &Ks[key * 72];
      const int cb = hiL ? 8 : 0;
      const v16h b0 = frag16(kr + cb, kr + cb + 16);
      const v16h b1 = frag16(kr + 32 + cb, kr + 32 + cb + 16);
      v8f t = __builtin_amdgcn_wmma_f32_16x16x32_f16(false, qa[0], false, b0,
                                                     (short)0, vzero, false, false);
      s[n] = __builtin_amdgcn_wmma_f32_16x16x32_f16(false, qa[1], false, b1,
                                                    (short)0, t, false, false);
    }

    // ---- causal mask on the diagonal key block ------------------------------
    if (kb == qblk) {
#pragma unroll
      for (int n = 0; n < 4; ++n)
#pragma unroll
        for (int r = 0; r < 8; ++r) {
          const int col = kb * 64 + n * 16 + lrow;
          const int row = qblk * 64 + wave * 16 + r + (hiL ? 8 : 0);
          if (col > row) s[n][r] = -1e30f;
        }
    }

    // ---- online softmax (row r lives across 16 lanes of one half) ----------
#pragma unroll
    for (int r = 0; r < 8; ++r) {
      float mv = fmaxf(fmaxf(s[0][r], s[1][r]), fmaxf(s[2][r], s[3][r]));
#pragma unroll
      for (int off = 1; off < 16; off <<= 1) mv = fmaxf(mv, __shfl_xor(mv, off, 32));
      const float mnew  = fmaxf(m_s[r], mv);
      const float alpha = __expf(m_s[r] - mnew);
      float rs = 0.f;
#pragma unroll
      for (int n = 0; n < 4; ++n) {
        const float p = __expf(s[n][r] - mnew);
        s[n][r] = p;
        rs += p;
      }
#pragma unroll
      for (int off = 1; off < 16; off <<= 1) rs += __shfl_xor(rs, off, 32);
      l_s[r] = l_s[r] * alpha + rs;
      m_s[r] = mnew;
#pragma unroll
      for (int n = 0; n < 4; ++n) yacc[n][r] *= alpha;
    }

    // ---- P: C-layout -> LDS -> A-layout fragments ---------------------------
    _Float16* pb = &Ps[wave * 16 * 72];
#pragma unroll
    for (int n = 0; n < 4; ++n)
#pragma unroll
      for (int r = 0; r < 8; ++r) {
        const int prow = r + (hiL ? 8 : 0);
        pb[prow * 72 + n * 16 + lrow] = (_Float16)s[n][r];
      }
    __syncthreads();

    // ---- Y += P * V : 2 key-chunks x 4 hd tiles = 8 WMMA --------------------
#pragma unroll
    for (int c = 0; c < 2; ++c) {
      const int off = c * 32 + (hiL ? 8 : 0);
      const v16h pa = frag16(&pb[lrow * 72 + off], &pb[lrow * 72 + off + 16]);
#pragma unroll
      for (int n = 0; n < 4; ++n) {
        const int hdc = n * 16 + lrow;
        const int cb  = hiL ? 8 : 0;
        const v16h vf = frag16(&Vs[hdc * 72 + c * 32 + cb],
                               &Vs[hdc * 72 + c * 32 + cb + 16]);
        yacc[n] = __builtin_amdgcn_wmma_f32_16x16x32_f16(false, pa, false, vf,
                                                         (short)0, yacc[n], false, false);
      }
    }
    __syncthreads();
  }

  // ---- finalize: y = yacc / l, scatter into [B][L][D] fp32 ------------------
#pragma unroll
  for (int n = 0; n < 4; ++n)
#pragma unroll
    for (int r = 0; r < 8; ++r) {
      const int row = qblk * 64 + wave * 16 + r + (hiL ? 8 : 0);
      const int col = h * kHD + n * 16 + lrow;
      yws[((size_t)b * kL + row) * kD + col] = yacc[n][r] / l_s[r];
    }
}

// ---------------------------------------------------------------------------
extern "C" void kernel_launch(void* const* d_in, const int* in_sizes, int n_in,
                              void* d_out, int out_size, void* d_ws, size_t ws_size,
                              hipStream_t stream) {
  (void)in_sizes; (void)n_in; (void)out_size; (void)ws_size;
  const float* x     = (const float*)d_in[0];
  const float* cosT  = (const float*)d_in[1];
  const float* sinT  = (const float*)d_in[2];
  const float* Wqkv  = (const float*)d_in[3];
  const float* Wproj = (const float*)d_in[4];
  float* out = (float*)d_out;

  char* ws = (char*)d_ws;
  size_t off = 0;
  float* qkvws = (float*)(ws + off);
  off += (size_t)3 * kB * kH * kL * kHD * sizeof(float);      // 100.7 MB
  _Float16* qh = (_Float16*)(ws + off);
  off += (size_t)kB * kH * kL * kHD * sizeof(_Float16);       // 16.8 MB
  _Float16* kh = (_Float16*)(ws + off);
  off += (size_t)kB * kH * kL * kHD * sizeof(_Float16);       // 16.8 MB
  _Float16* vth = (_Float16*)(ws + off);
  off += (size_t)kB * kH * kL * kHD * sizeof(_Float16);       // 16.8 MB
  float* yws = (float*)(ws + off);
  off += (size_t)kB * kL * kD * sizeof(float);                // 33.6 MB

  // 1) qkv = x @ Wqkv^T  (scattered into [3][B][H][L][HD])
  gemm_wmma<1><<<dim3(3 * kD / 128, kM / 128), 256, 0, stream>>>(
      x, Wqkv, qkvws, 3 * kD, kD);
  // 2) RoPE + f16 pack (q scaled by 1/sqrt(HD)), v transposed
  rope_pack<<<dim3(kB * kH * kL / 4), 256, 0, stream>>>(qkvws, cosT, sinT, qh, kh, vth);
  // 3) causal flash attention
  attn_wmma<<<dim3(kL / 64, kH, kB), 128, 0, stream>>>(qh, kh, vth, yws);
  // 4) out = y @ Wproj^T
  gemm_wmma<0><<<dim3(kD / 128, kM / 128), 256, 0, stream>>>(yws, Wproj, out, kD, kD);
}
